// T5AttentionMod_1357209666476
// MI455X (gfx1250) — matrix operand order repaired and checked by reading, not compiled, not run on hardware
//
#include <hip/hip_runtime.h>
#include <cstdint>

typedef __attribute__((ext_vector_type(16))) _Float16 v16h;
typedef __attribute__((ext_vector_type(8)))  _Float16 v8h;
typedef __attribute__((ext_vector_type(8)))  float    v8f;

#define S_LEN 2048
#define HD    64
#define NH    16
#define DMODEL 1024

// ---------------------------------------------------------------------------
// CDNA5 async copy helpers: global -> LDS DMA (no VGPR round trip), tracked
// by ASYNCcnt.  LDS destination passed as 32-bit addrspace(3) offset.
// ---------------------------------------------------------------------------
__device__ __forceinline__ uint32_t lds_off_of(const void* p) {
  return (uint32_t)(uintptr_t)(const __attribute__((address_space(3))) char*)p;
}

// 16-byte chunk: LDS[dst] = *(b128*)(gbase + 2*elem_off)
__device__ __forceinline__ void async_ld_b128(const void* lds_dst,
                                              const _Float16* gbase,
                                              uint32_t elem_off) {
  uint32_t l = lds_off_of(lds_dst);
  uint64_t b = (uint64_t)(uintptr_t)gbase;
  uint32_t v = elem_off * 2u;   // byte offset from SGPR base
  asm volatile("global_load_async_to_lds_b128 %0, %1, %2"
               :: "v"(l), "v"(v), "s"(b) : "memory");
}

__device__ __forceinline__ void wait_async0() {
  asm volatile("s_wait_asynccnt 0" ::: "memory");
}

// ---------------------------------------------------------------------------
// GEMM: C[M,N] = A[M,K] * W[K,N].  256 threads = 8 waves, 128x128 tile/block.
// Wave grid 4x2: each wave owns 32 rows x 64 cols = 2x4 WMMA accumulators,
// 8 x v_wmma_f32_16x16x32_f16 per 32-wide K step (12 ds_load_b128 / 8 WMMA).
// A staged f16 row-major in LDS (async DMA when A is already f16);
// W staged f16 transposed [N][K] so B-fragments are contiguous v16h reads.
// OUT_MODE: 0 = f32 row-major [M,N] (final output, non-temporal stores)
//           1 = f16 split-head [B,H,S,hd]            (Q,K path)
//           2 = f16 split-head transposed [B,H,hd,S] (V path -> P@V B-frag)
// ---------------------------------------------------------------------------
template<bool A_F16, int OUT_MODE>
__global__ void __launch_bounds__(256)
gemm128(const void* __restrict__ Av, const float* __restrict__ W,
        void* __restrict__ Cv, int M, int N, int K)
{
  __shared__ __align__(32) _Float16 As[128][48];   // [m][k], stride 96B
  __shared__ __align__(32) _Float16 Ws[128][48];   // [n][k] (transposed W tile)

  const int t = threadIdx.x;
  const int lane = t & 31, wave = t >> 5;           // 8 waves
  const int lanelo = lane & 15, hi = lane >> 4, hi8 = hi * 8;
  const int wm = wave >> 1, wn = wave & 1;          // 4 x 2 wave grid
  const int m0 = blockIdx.y * 128, n0 = blockIdx.x * 128;

  v8f acc[2][4] = {};

  for (int k0 = 0; k0 < K; k0 += 32) {
    // ---- stage A tile 128x32 ----
    if constexpr (A_F16) {
      // already f16 row-major: pure async DMA, 512 x 16B chunks, 2/thread
      const _Float16* Ah = (const _Float16*)Av;
#pragma unroll
      for (int j = 0; j < 2; ++j) {
        const int ci = t + 256 * j;
        const int r = ci >> 2, c8 = (ci & 3) * 8;
        async_ld_b128(&As[r][c8], Ah,
                      (uint32_t)((m0 + r) * K + k0 + c8));
      }
    } else {
      // f32 -> f16 conversion through VGPRs, float4 vector loads
      const float* Af = (const float*)Av;
#pragma unroll
      for (int j = 0; j < 4; ++j) {
        const int q = t + 256 * j;
        const int r = q >> 3, c4 = (q & 7) * 4;
        const float4 f = *(const float4*)&Af[(size_t)(m0 + r) * K + k0 + c4];
        As[r][c4 + 0] = (_Float16)f.x;
        As[r][c4 + 1] = (_Float16)f.y;
        As[r][c4 + 2] = (_Float16)f.z;
        As[r][c4 + 3] = (_Float16)f.w;
      }
    }
    // ---- stage W tile 32x128 transposed into [n][k], float4 loads ----
#pragma unroll
    for (int j = 0; j < 4; ++j) {
      const int q = t + 256 * j;
      const int kk = q >> 5, n4 = (q & 31) * 4;
      const float4 f = *(const float4*)&W[(size_t)(k0 + kk) * N + n0 + n4];
      Ws[n4 + 0][kk] = (_Float16)f.x;
      Ws[n4 + 1][kk] = (_Float16)f.y;
      Ws[n4 + 2][kk] = (_Float16)f.z;
      Ws[n4 + 3][kk] = (_Float16)f.w;
    }
    if constexpr (A_F16) wait_async0();
    __syncthreads();

    // A fragments: lane<16 = row, K 0-15; lane>=16 = K 16-31
    v16h a[2];
#pragma unroll
    for (int mt = 0; mt < 2; ++mt) {
      const int arow = wm * 32 + mt * 16 + lanelo;
      v8h lo = *(const v8h*)&As[arow][hi * 16];
      v8h hv = *(const v8h*)&As[arow][hi * 16 + 8];
#pragma unroll
      for (int i = 0; i < 8; ++i) { a[mt][i] = lo[i]; a[mt][i + 8] = hv[i]; }
    }
#pragma unroll
    for (int nt = 0; nt < 4; ++nt) {
      // B fragment: lane<16 = col, K 0-15; lane>=16 = col, K 16-31
      v16h bfr = *(const v16h*)&Ws[wn * 64 + nt * 16 + lanelo][hi * 16];
#pragma unroll
      for (int mt = 0; mt < 2; ++mt)
        acc[mt][nt] = __builtin_amdgcn_wmma_f32_16x16x32_f16(
            false, a[mt], false, bfr, (short)0, acc[mt][nt], false, false);
    }
    __syncthreads();
  }

  // C fragment: VGPR r -> row (hi8 + r), col lanelo within each 16x16 tile
#pragma unroll
  for (int mt = 0; mt < 2; ++mt)
#pragma unroll
    for (int nt = 0; nt < 4; ++nt)
#pragma unroll
      for (int r = 0; r < 8; ++r) {
        const int gm = m0 + wm * 32 + mt * 16 + hi8 + r;
        const int gn = n0 + wn * 64 + nt * 16 + lanelo;
        const float v = acc[mt][nt][r];
        const int b = gm >> 11, s = gm & (S_LEN - 1);   // M = B*S, S=2048
        const int h = gn >> 6, hd = gn & (HD - 1);
        if constexpr (OUT_MODE == 1) {
          ((_Float16*)Cv)[(((size_t)(b * NH + h)) * S_LEN + s) * HD + hd] =
              (_Float16)v;
        } else if constexpr (OUT_MODE == 2) {
          ((_Float16*)Cv)[(((size_t)(b * NH + h)) * HD + hd) * S_LEN + s] =
              (_Float16)v;
        } else {
          __builtin_nontemporal_store(v, &((float*)Cv)[(size_t)gm * N + gn]);
        }
      }
}

// ---------------------------------------------------------------------------
// Fused flash-style attention.  One 128-thread block per (b, h, 64 q-rows).
// Q resident in LDS; 32 key-blocks of 64, K/V staged with double-buffered
// GLOBAL_LOAD_ASYNC_TO_LDS_B128 (ASYNCcnt) so block kb+1 streams in while
// WMMAs run on block kb.  V arrives pre-transposed [hd][S] from its GEMM so
// both K and V tiles are pure contiguous b128 DMA (no transpose, no VGPRs).
//   scores = Q @ K^T + bias(NT loads)  ->  online softmax (shfl_xor)
//   ctx   += P @ V  (P via per-wave LDS tile + s_wait_dscnt 0)
// ---------------------------------------------------------------------------
__global__ void __launch_bounds__(128)
attn_kernel(const _Float16* __restrict__ Q, const _Float16* __restrict__ Kg,
            const _Float16* __restrict__ Vg, const float* __restrict__ bias,
            _Float16* __restrict__ ctx)
{
  __shared__ __align__(32) _Float16 Qs[64][80];        // [q][hd]     stride 160B
  __shared__ __align__(32) _Float16 Ks[2][64][80];     // [key][hd]   B-frag layout
  __shared__ __align__(32) _Float16 Vt[2][64][80];     // [hd][key]   B-frag layout
  __shared__ __align__(32) _Float16 Ps[4][16][80];     // per-wave P tile [q][key]

  const int t = threadIdx.x, lane = t & 31, wave = t >> 5;
  const int lanelo = lane & 15, hi = lane >> 4, hi8 = hi * 8;

  const int bid = blockIdx.x;
  const int qb = bid & 31;             // S/64 = 32 q-blocks
  const int h  = (bid >> 5) & (NH - 1);
  const int b  = bid >> 9;
  const int q0 = qb * 64;
  const uint32_t headoff = (uint32_t)(b * NH + h) * (S_LEN * HD); // elements

  // chunk decomposition: 64x64 f16 tile = 512 x 16B chunks, 4 per thread
  const int ci0 = t;                   // + 128*j
  // ---- issue async DMA: Q block + K/V block 0 --------------------------
#pragma unroll
  for (int j = 0; j < 4; ++j) {
    const int ci = ci0 + 128 * j;
    const int r = ci >> 3, c8 = (ci & 7) * 8;
    async_ld_b128(&Qs[r][c8],    Q,  headoff + (uint32_t)(q0 + r) * HD + c8);
    async_ld_b128(&Ks[0][r][c8], Kg, headoff + (uint32_t)r * HD + c8);
    async_ld_b128(&Vt[0][r][c8], Vg, headoff + (uint32_t)r * S_LEN + c8);
  }
  wait_async0();
  __syncthreads();

  // Q fragments for this wave's 16 rows (persist across the whole loop)
  const int arow = wave * 16 + lanelo;
  const int koff = hi * 16;
  v16h qa[2];
#pragma unroll
  for (int ks = 0; ks < 2; ++ks) {
    v8h lo = *(const v8h*)&Qs[arow][ks * 32 + koff];
    v8h hv = *(const v8h*)&Qs[arow][ks * 32 + koff + 8];
#pragma unroll
    for (int i = 0; i < 8; ++i) { qa[ks][i] = lo[i]; qa[ks][i + 8] = hv[i]; }
  }

  float mrow[8], srow[8];
#pragma unroll
  for (int r = 0; r < 8; ++r) { mrow[r] = -3.0e38f; srow[r] = 0.0f; }
  v8f O[4] = {};

  const float* bp = bias + ((size_t)h * S_LEN + q0) * S_LEN;

  for (int kb = 0; kb < S_LEN / 64; ++kb) {
    const int cur = kb & 1;
    // prefetch next block into the other buffer (overlaps with compute below)
    if (kb + 1 < S_LEN / 64) {
      const uint32_t kgoff = headoff + (uint32_t)(kb + 1) * 64 * HD; // [key][hd]
      const uint32_t vgoff = headoff + (uint32_t)(kb + 1) * 64;      // [hd][S]
#pragma unroll
      for (int j = 0; j < 4; ++j) {
        const int ci = ci0 + 128 * j;
        const int r = ci >> 3, c8 = (ci & 7) * 8;
        async_ld_b128(&Ks[cur ^ 1][r][c8], Kg, kgoff + (uint32_t)r * HD + c8);
        async_ld_b128(&Vt[cur ^ 1][r][c8], Vg, vgoff + (uint32_t)r * S_LEN + c8);
      }
    }

    // ---- scores: 4 key-tiles x (2 WMMA over hd=64) ----
    v8f sc[4];
#pragma unroll
    for (int kt = 0; kt < 4; ++kt) {
      v8f z = {};
#pragma unroll
      for (int ks = 0; ks < 2; ++ks) {
        v16h bf = *(const v16h*)&Ks[cur][kt * 16 + lanelo][ks * 32 + hi * 16];
        z = __builtin_amdgcn_wmma_f32_16x16x32_f16(
            false, qa[ks], false, bf, (short)0, z, false, false);
      }
      sc[kt] = z;
    }
    // additive position bias, streamed once -> non-temporal loads
#pragma unroll
    for (int kt = 0; kt < 4; ++kt)
#pragma unroll
      for (int r = 0; r < 8; ++r)
        sc[kt][r] += __builtin_nontemporal_load(
            &bp[(size_t)(wave * 16 + hi8 + r) * S_LEN
                + kb * 64 + kt * 16 + lanelo]);

    // ---- online softmax per q-row ----
#pragma unroll
    for (int r = 0; r < 8; ++r) {
      float v = fmaxf(fmaxf(sc[0][r], sc[1][r]), fmaxf(sc[2][r], sc[3][r]));
#pragma unroll
      for (int mk = 1; mk < 16; mk <<= 1)
        v = fmaxf(v, __shfl_xor(v, mk, 32));
      const float mn = fmaxf(mrow[r], v);
      const float scale = __expf(mrow[r] - mn);
      mrow[r] = mn;
      srow[r] *= scale;
#pragma unroll
      for (int nt = 0; nt < 4; ++nt) O[nt][r] *= scale;
      float ps = 0.0f;
#pragma unroll
      for (int kt = 0; kt < 4; ++kt) {
        const float p = __expf(sc[kt][r] - mn);
        sc[kt][r] = p;
        ps += p;
      }
#pragma unroll
      for (int mk = 1; mk < 16; mk <<= 1)
        ps += __shfl_xor(ps, mk, 32);
      srow[r] += ps;
    }

    // ---- P -> per-wave LDS tile (f16); wave-internal DS hazard wait ----
#pragma unroll
    for (int kt = 0; kt < 4; ++kt)
#pragma unroll
      for (int r = 0; r < 8; ++r)
        Ps[wave][hi8 + r][kt * 16 + lanelo] = (_Float16)sc[kt][r];
    asm volatile("s_wait_dscnt 0" ::: "memory");

    // ---- ctx += P @ V (K-dim = 64 keys -> 2 steps of 32) ----
#pragma unroll
    for (int ks = 0; ks < 2; ++ks) {
      v8h lo = *(const v8h*)&Ps[wave][lanelo][ks * 32 + koff];
      v8h hv = *(const v8h*)&Ps[wave][lanelo][ks * 32 + koff + 8];
      v16h pa;
#pragma unroll
      for (int i = 0; i < 8; ++i) { pa[i] = lo[i]; pa[i + 8] = hv[i]; }
#pragma unroll
      for (int nt = 0; nt < 4; ++nt) {
        v16h bf = *(const v16h*)&Vt[cur][nt * 16 + lanelo][ks * 32 + hi * 16];
        O[nt] = __builtin_amdgcn_wmma_f32_16x16x32_f16(
            false, pa, false, bf, (short)0, O[nt], false, false);
      }
    }

    // drain our DMA for the next buffer, then rendezvous: after the barrier
    // every wave's copies are visible and no wave still reads buffer cur.
    wait_async0();
    __syncthreads();
  }

  // normalize and emit ctx (f16, merged-head [B,S,D]) for the out-proj GEMM
#pragma unroll
  for (int r = 0; r < 8; ++r) {
    const float inv = 1.0f / srow[r];
    const int gq = q0 + wave * 16 + hi8 + r;
    const size_t base = ((size_t)b * S_LEN + gq) * DMODEL + h * HD;
#pragma unroll
    for (int nt = 0; nt < 4; ++nt)
      ctx[base + nt * 16 + lanelo] = (_Float16)(O[nt][r] * inv);
  }
}

// ---------------------------------------------------------------------------
extern "C" void kernel_launch(void* const* d_in, const int* in_sizes, int n_in,
                              void* d_out, int out_size, void* d_ws, size_t ws_size,
                              hipStream_t stream) {
  const float* hidden = (const float*)d_in[0];   // [2,2048,1024]
  const float* pbias  = (const float*)d_in[1];   // [1,16,2048,2048]
  const float* Wq     = (const float*)d_in[2];   // [1024,1024]
  const float* Wk     = (const float*)d_in[3];
  const float* Wv     = (const float*)d_in[4];
  const float* Wo     = (const float*)d_in[5];
  float* out = (float*)d_out;

  const int M = 2 * S_LEN;         // 4096
  const int N = DMODEL;            // 1024
  const int K = DMODEL;            // 1024

  const size_t mat = (size_t)M * DMODEL;         // 4M f16 elements
  _Float16* qws = (_Float16*)d_ws;
  _Float16* kws = qws + mat;
  _Float16* vws = kws + mat;                     // [B,H,hd,S] (transposed)
  _Float16* cws = vws + mat;                     // total 32 MB workspace

  dim3 gg(N / 128, M / 128);       // (8, 32)
  dim3 bb(256);

  gemm128<false, 1><<<gg, bb, 0, stream>>>(hidden, Wq, qws, M, N, K);
  gemm128<false, 1><<<gg, bb, 0, stream>>>(hidden, Wk, kws, M, N, K);
  gemm128<false, 2><<<gg, bb, 0, stream>>>(hidden, Wv, vws, M, N, K);

  attn_kernel<<<dim3(2 * NH * (S_LEN / 64)), 128, 0, stream>>>(
      qws, kws, vws, pbias, cws);

  gemm128<true, 0><<<gg, bb, 0, stream>>>(cws, Wo, out, M, N, K);
}
